// MultiHeadAttentionWithRoPE_16423954940361
// MI455X (gfx1250) — compile-verified, matrix operand order checked by reading, not run on hardware
//
#include <hip/hip_runtime.h>

typedef __attribute__((ext_vector_type(16))) __bf16 v16bf;
typedef __attribute__((ext_vector_type(8)))  __bf16 bf16x8;
typedef __attribute__((ext_vector_type(8)))  float  v8f;

__device__ __forceinline__ v8f wmma_bf16(v16bf a, v16bf b, v8f c) {
  return __builtin_amdgcn_wmma_f32_16x16x32_bf16(false, a, false, b, (short)0, c,
                                                 false, false);
}

// Fragment gather per CDNA5 ISA 7.12.2 for unit-stride K: each lane's 16
// bf16 values are two contiguous 16B runs -> two b128 loads.
// rowStride in elements; (rowStride*2 B), kOff*2 B and base must be 16B-aligned.
__device__ __forceinline__ v16bf load_frag_k1(const __bf16* base, int row0,
                                              int rowStride, int kOff) {
  const int lane = threadIdx.x & 31;
  const int r = lane & 15, half = lane >> 4;
  const __bf16* p = base + (size_t)(row0 + r) * rowStride + kOff + half * 8;
  const bf16x8 lo = *(const bf16x8*)p;
  const bf16x8 hi = *(const bf16x8*)(p + 16);
  return __builtin_shufflevector(lo, hi, 0, 1, 2, 3, 4, 5, 6, 7,
                                 8, 9, 10, 11, 12, 13, 14, 15);
}

// 4-element staging copy into LDS (converts f32 -> bf16, passthrough bf16).
__device__ __forceinline__ void cp4(const float* __restrict__ s,
                                    __bf16* __restrict__ d) {
  const float4 v = *(const float4*)s;
  union { __bf16 b[4]; unsigned long long u; } pk;
  pk.b[0] = (__bf16)v.x; pk.b[1] = (__bf16)v.y;
  pk.b[2] = (__bf16)v.z; pk.b[3] = (__bf16)v.w;
  *(unsigned long long*)d = pk.u;
}
__device__ __forceinline__ void cp4(const __bf16* __restrict__ s,
                                    __bf16* __restrict__ d) {
  *(unsigned long long*)d = *(const unsigned long long*)s;
}

// ---------------------------------------------------------------------------
// Tiled NT GEMM: C[M,N] = A[M,K] * B[N,K]^T (+ bias), bf16 WMMA, f32 accum.
// 256 threads = 8 waves, 128x128 tile, BK=32, LDS rows 80B (16B-aligned).
// ---------------------------------------------------------------------------
template <typename TA, typename TB, typename TO, bool HAS_BIAS>
__global__ __launch_bounds__(256) void gemm_nt_kernel(
    const TA* __restrict__ A, const TB* __restrict__ Bm,
    const float* __restrict__ bias, TO* __restrict__ C, int M, int N, int K) {
  constexpr int BM = 128, BN = 128, BK = 32, LDK = 40;
  __shared__ alignas(16) __bf16 sA[BM * LDK];
  __shared__ alignas(16) __bf16 sB[BN * LDK];
  const int bm = blockIdx.x * BM;
  const int bn = blockIdx.y * BN;
  const int tid = threadIdx.x;
  const int wid = tid >> 5, lane = tid & 31;
  const int wm = (wid & 3) * 32;   // 4 waves over M
  const int wn = (wid >> 2) * 64;  // 2 waves over N
  v8f acc[2][4] = {};
  for (int k0 = 0; k0 < K; k0 += BK) {
#pragma unroll
    for (int i = tid; i < BM * BK / 4; i += 256) {  // 4 iters/thread
      const int r = i >> 3, c = (i & 7) * 4;
      cp4(&A[(size_t)(bm + r) * K + k0 + c], &sA[r * LDK + c]);
      cp4(&Bm[(size_t)(bn + r) * K + k0 + c], &sB[r * LDK + c]);
    }
    __syncthreads();
    const v16bf af0 = load_frag_k1(sA, wm, LDK, 0);
    const v16bf af1 = load_frag_k1(sA, wm + 16, LDK, 0);
#pragma unroll
    for (int ni = 0; ni < 4; ++ni) {
      const v16bf bfg = load_frag_k1(sB, wn + ni * 16, LDK, 0);
      acc[0][ni] = wmma_bf16(af0, bfg, acc[0][ni]);
      acc[1][ni] = wmma_bf16(af1, bfg, acc[1][ni]);
    }
    __syncthreads();
  }
  const int half = lane >> 4, col = lane & 15;
#pragma unroll
  for (int mi = 0; mi < 2; ++mi)
#pragma unroll
    for (int ni = 0; ni < 4; ++ni)
#pragma unroll
      for (int r = 0; r < 8; ++r) {
        const int row = bm + wm + mi * 16 + r + half * 8;
        const int cn  = bn + wn + ni * 16 + col;
        float v = acc[mi][ni][r];
        if (HAS_BIAS) v += bias[cn];
        C[(size_t)row * N + cn] = (TO)v;
      }
}

// ---------------------------------------------------------------------------
// RoPE (base=50) on q,k + reorder. Q,K -> [B,H,S,dk]; V -> transposed
// [B,H,dk,S] so the PV matmul's B-fragments are unit-stride in K.
// ---------------------------------------------------------------------------
__global__ __launch_bounds__(256) void rope_reorder_kernel(
    const __bf16* __restrict__ qkv, __bf16* __restrict__ Q,
    __bf16* __restrict__ K, __bf16* __restrict__ Vt) {
  const int idx = blockIdx.x * 256 + threadIdx.x;
  const int i = idx & 31;          // pair index within dk (dk/2 = 32)
  int t = idx >> 5;
  const int h = t & 15; t >>= 4;
  const int s = t & 2047;
  const int b = t >> 11;
  const size_t src = (size_t)(b * 2048 + s) * 3072 + h * 64 + 2 * i;
  const float q0 = (float)qkv[src],        q1 = (float)qkv[src + 1];
  const float k0 = (float)qkv[src + 1024], k1 = (float)qkv[src + 1025];
  const float v0 = (float)qkv[src + 2048], v1 = (float)qkv[src + 2049];
  const float freq = __expf(-((float)(2 * i) * (1.0f / 64.0f)) * __logf(50.0f));
  const float ang = (float)s * freq;
  float sn, cs;
  __sincosf(ang, &sn, &cs);
  const size_t dst = (size_t)((b * 16 + h) * 2048 + s) * 64 + 2 * i;
  Q[dst]     = (__bf16)(q0 * cs - q1 * sn);
  Q[dst + 1] = (__bf16)(q1 * cs + q0 * sn);
  K[dst]     = (__bf16)(k0 * cs - k1 * sn);
  K[dst + 1] = (__bf16)(k1 * cs + k0 * sn);
  const size_t vrow = (size_t)((b * 16 + h) * 64 + 2 * i) * 2048 + s;
  Vt[vrow]        = (__bf16)v0;
  Vt[vrow + 2048] = (__bf16)v1;
}

// ---------------------------------------------------------------------------
// Flash-style attention with diagonal mask. Block = 128 threads = 4 waves.
// Each wave owns 16 query rows x dk=64 output; KV streamed in 128-key chunks.
// ---------------------------------------------------------------------------
#define CT 128
__global__ __launch_bounds__(128) void attn_kernel(
    const __bf16* __restrict__ Q, const __bf16* __restrict__ Kc,
    const __bf16* __restrict__ Vt, __bf16* __restrict__ Out) {
  const int S = 2048, dk = 64, H = 16, D = 1024;
  int bid = blockIdx.x;
  const int rowBlk = bid & 31; bid >>= 5;   // S/64 = 32 row-blocks
  const int h = bid & 15;
  const int b = bid >> 4;
  __shared__ alignas(16) __bf16 sK[CT * 64];     // [t][d]
  __shared__ alignas(16) __bf16 sVt[64 * CT];    // [d][t]
  __shared__ alignas(16) float  sS[4][16 * CT];
  __shared__ alignas(16) __bf16 sP[4][16 * CT];
  __shared__ float  sAlpha[4][16];
  __shared__ float  sL[4][16];
  const int tid = threadIdx.x, wid = tid >> 5, lane = tid & 31;
  const int half = lane >> 4, cidx = lane & 15;
  const int qrow0 = rowBlk * 64 + wid * 16;
  const __bf16* qbase  = Q  + (size_t)(b * H + h) * S * dk;
  const __bf16* kbase  = Kc + (size_t)(b * H + h) * S * dk;
  const __bf16* vtbase = Vt + (size_t)(b * H + h) * dk * S;
  const v16bf aq0 = load_frag_k1(qbase, qrow0, dk, 0);
  const v16bf aq1 = load_frag_k1(qbase, qrow0, dk, 32);
  v8f o[4] = {};
  float mstate = -1e30f, lstate = 0.0f;

  for (int t0 = 0; t0 < S; t0 += CT) {
    // Cooperative KV chunk staging (16B vector copies).
    const uint4* kg = (const uint4*)(kbase + (size_t)t0 * dk);  // contiguous
    uint4* skp = (uint4*)sK;
#pragma unroll
    for (int i = tid; i < CT * dk / 8; i += 128) skp[i] = kg[i];  // 8 iters
    const uint4* vg = (const uint4*)vtbase;
    uint4* svp = (uint4*)sVt;
#pragma unroll
    for (int i = tid; i < dk * CT / 8; i += 128) {                // 8 iters
      const int row = i >> 4, c4 = i & 15;  // 16 uint4 per 128-col row
      svp[row * 16 + c4] = vg[row * (S / 8) + (t0 / 8) + c4];
    }
    __syncthreads();

    // scores = Q*K^T * 1/sqrt(dk) for this wave's 16 rows x CT cols
#pragma unroll
    for (int nt = 0; nt < 8; ++nt) {
      const v16bf b0 = load_frag_k1(sK, nt * 16, dk, 0);
      const v16bf b1 = load_frag_k1(sK, nt * 16, dk, 32);
      v8f c = {};
      c = wmma_bf16(aq0, b0, c);
      c = wmma_bf16(aq1, b1, c);
#pragma unroll
      for (int r = 0; r < 8; ++r)
        sS[wid][(r + half * 8) * CT + nt * 16 + cidx] = c[r] * 0.125f;
    }
    asm volatile("s_wait_dscnt 0" ::: "memory");  // wave-local LDS RAW

    // online softmax: 2 lanes per row, 64 cols each (vectorized)
    {
      const int row = lane >> 1;
      const int seg = (lane & 1) * 64;
      const int grow = qrow0 + row;
      float* srow = &sS[wid][row * CT + seg];
      const int dcol = grow - t0 - seg;            // diagonal mask
      if (dcol >= 0 && dcol < 64) srow[dcol] = -1e30f;
      const float4* srow4 = (const float4*)srow;
      float mx = -1e30f;
#pragma unroll
      for (int c4 = 0; c4 < 16; ++c4) {
        const float4 v = srow4[c4];
        mx = fmaxf(mx, fmaxf(fmaxf(v.x, v.y), fmaxf(v.z, v.w)));
      }
      mx = fmaxf(mx, __shfl_xor(mx, 1, 32));
      const float mnew  = fmaxf(mstate, mx);
      const float alpha = __expf(mstate - mnew);
      float ssum = 0.0f;
      unsigned long long* prow4 =
          (unsigned long long*)&sP[wid][row * CT + seg];
#pragma unroll
      for (int c4 = 0; c4 < 16; ++c4) {
        const float4 v = srow4[c4];
        const float p0 = __expf(v.x - mnew), p1 = __expf(v.y - mnew);
        const float p2 = __expf(v.z - mnew), p3 = __expf(v.w - mnew);
        ssum += (p0 + p1) + (p2 + p3);
        union { __bf16 b[4]; unsigned long long u; } pk;
        pk.b[0] = (__bf16)p0; pk.b[1] = (__bf16)p1;
        pk.b[2] = (__bf16)p2; pk.b[3] = (__bf16)p3;
        prow4[c4] = pk.u;
      }
      ssum += __shfl_xor(ssum, 1, 32);
      lstate = lstate * alpha + ssum;
      mstate = mnew;
      if ((lane & 1) == 0) { sAlpha[wid][row] = alpha; sL[wid][row] = lstate; }
    }
    asm volatile("s_wait_dscnt 0" ::: "memory");

    // rescale accumulators
    float al[8];
#pragma unroll
    for (int r = 0; r < 8; ++r) al[r] = sAlpha[wid][r + half * 8];
#pragma unroll
    for (int nt = 0; nt < 4; ++nt)
#pragma unroll
      for (int r = 0; r < 8; ++r) o[nt][r] *= al[r];

    // O += P * V   (P: 16xCT; V^T staged as [d][t] -> unit-stride K frags)
#pragma unroll
    for (int kt = 0; kt < 4; ++kt) {
      const v16bf ap = load_frag_k1(sP[wid], 0, CT, kt * 32);
#pragma unroll
      for (int nt = 0; nt < 4; ++nt) {
        const v16bf bv = load_frag_k1(sVt, nt * 16, CT, kt * 32);
        o[nt] = wmma_bf16(ap, bv, o[nt]);
      }
    }
    __syncthreads();
  }

  float li[8];
#pragma unroll
  for (int r = 0; r < 8; ++r) li[r] = 1.0f / sL[wid][r + half * 8];
#pragma unroll
  for (int nt = 0; nt < 4; ++nt)
#pragma unroll
    for (int r = 0; r < 8; ++r) {
      const int grow = qrow0 + r + half * 8;
      const int d = nt * 16 + cidx;
      Out[(size_t)(b * S + grow) * D + h * dk + d] = (__bf16)(o[nt][r] * li[r]);
    }
}

// ---------------------------------------------------------------------------
extern "C" void kernel_launch(void* const* d_in, const int* in_sizes, int n_in,
                              void* d_out, int out_size, void* d_ws, size_t ws_size,
                              hipStream_t stream) {
  const float* x     = (const float*)d_in[0];
  const float* w_qkv = (const float*)d_in[1];
  const float* w_out = (const float*)d_in[2];
  const float* b_out = (const float*)d_in[3];
  float* out = (float*)d_out;

  const int B = 2, S = 2048, D = 1024, H = 16, dk = 64;
  const int M = B * S;  // 4096

  char* ws = (char*)d_ws;
  __bf16* qkv = (__bf16*)ws;                                  // M * 3D bf16
  __bf16* Qp  = (__bf16*)(ws + (size_t)M * 3 * D * 2);
  __bf16* Kp  = Qp + (size_t)B * H * S * dk;
  __bf16* Vtp = Kp + (size_t)B * H * S * dk;                  // [B,H,dk,S]
  __bf16* ctx = Vtp + (size_t)B * H * dk * S;                 // [B,S,D]

  gemm_nt_kernel<float, float, __bf16, false>
      <<<dim3(M / 128, (3 * D) / 128), 256, 0, stream>>>(x, w_qkv, nullptr, qkv,
                                                         M, 3 * D, D);
  rope_reorder_kernel<<<(B * S * H * (dk / 2)) / 256, 256, 0, stream>>>(qkv, Qp,
                                                                        Kp, Vtp);
  attn_kernel<<<B * H * (S / 64), 128, 0, stream>>>(Qp, Kp, Vtp, ctx);
  gemm_nt_kernel<__bf16, float, float, true>
      <<<dim3(M / 128, D / 128), 256, 0, stream>>>(ctx, w_out, b_out, out,
                                                   M, D, D);
}
